// MultiheadLatentAttention_45226005626959
// MI455X (gfx1250) — matrix operand order repaired and checked by reading, not compiled, not run on hardware
//
#include <hip/hip_runtime.h>
#include <math.h>

// ---------------------------------------------------------------------------
// MLA for MI455X / gfx1250, wave32 + WMMA, bf16-resident pipeline.
//   * One-time f32->bf16 conversion (+pre-transpose of B operands) kernels.
//   * One generic batched GEMM: all operands bf16, K-contiguous [M,K]x[N,K],
//     f32 accumulation in v_wmma_f32_16x16x32_bf16, bf16 (or f32) C store.
//   * Tile staging uses CDNA5 async VMEM->LDS copies
//     (global_load_async_to_lds_b128 + s_wait_asynccnt), zero VGPR traffic.
//   * RoPE / softmax are bf16-storage, f32-math elementwise kernels.
// ---------------------------------------------------------------------------

typedef __bf16 bf16_t;
typedef __attribute__((ext_vector_type(16))) __bf16          v16bf;
typedef __attribute__((ext_vector_type(8)))  float           v8f;
typedef __attribute__((ext_vector_type(4)))  float           f32x4;
typedef __attribute__((ext_vector_type(4)))  unsigned short  u16x4;
typedef __attribute__((ext_vector_type(8)))  unsigned short  u16x8;
typedef __attribute__((ext_vector_type(16))) unsigned short  u16x16;

__device__ __forceinline__ unsigned short f32_to_bf16_bits(float f) {
    union { float f; unsigned u; } v; v.f = f;
    unsigned u = v.u + 0x7FFFu + ((v.u >> 16) & 1u);   // round-to-nearest-even
    return (unsigned short)(u >> 16);
}
__device__ __forceinline__ float bf16_bits_to_f32(unsigned short h) {
    union { unsigned u; float f; } v; v.u = ((unsigned)h) << 16;
    return v.f;
}

// Concat two 16B LDS chunks into one 32B bf16x16 fragment (register-only).
__device__ __forceinline__ v16bf make_frag(u16x8 lo, u16x8 hi) {
    u16x16 full = __builtin_shufflevector(lo, hi,
        0, 1, 2, 3, 4, 5, 6, 7, 8, 9, 10, 11, 12, 13, 14, 15);
    return __builtin_bit_cast(v16bf, full);
}

// CDNA5 async global->LDS copy, 16B per lane. lds_off is the wave-relative
// LDS byte address (low 32 bits of the flat LDS pointer). Tracked by ASYNCcnt.
__device__ __forceinline__ void async_copy_b128(unsigned lds_off,
                                                const unsigned short* g) {
    asm volatile("global_load_async_to_lds_b128 %0, %1, off"
                 :: "v"(lds_off), "v"(g)
                 : "memory");
}
__device__ __forceinline__ void wait_asynccnt0() {
    asm volatile("s_wait_asynccnt 0" ::: "memory");
}
__device__ __forceinline__ unsigned lds_addr_of(const void* p) {
    return (unsigned)(unsigned long long)p;   // addrspace(3) offset in low 32b
}

// ---------------------------------------------------------------------------
// Generic batched bf16 GEMM:  C[z] = A[z] * Bt[z]^T
//   A : [M,Kd] bf16 row-major (lda row stride)
//   Bt: [N,Kd] bf16 row-major (ldb row stride)  — K-contiguous for both.
// Macro-tile 256(M) x 64(N), K-step 32; 256 threads = 8 waves, wave w owns
// rows [w*32, w*32+32) (2 A fragments) and all 64 columns (4 B fragments)
// -> 8 WMMAs per K-step, each B fragment reused twice.
// z = (bb, hh) = (z / Hz, z % Hz) with independent element strides.
// All dims are multiples of the tile sizes for every call below.
// ---------------------------------------------------------------------------
#define BM 256
#define BN 64
#define BK 32

template<bool CF32>
__global__ __launch_bounds__(256)
void gemm_bf16_wmma(const unsigned short* __restrict__ A,
                    const unsigned short* __restrict__ Bt,
                    void* __restrict__ C,
                    int Kd, int lda, int ldb, int ldc,
                    long long sAb, long long sAh,
                    long long sBb, long long sBh,
                    long long sCb, long long sCh,
                    int Hz)
{
    // A tile: row-major [256][32] bf16.  B tile: fragment-swizzled
    // [4 subtiles][32 lanes][16 elems] bf16.
    __shared__ unsigned short Asl[BM * BK];          // 16 KB
    __shared__ unsigned short Bswz[4 * 32 * 16];     // 4 KB

    const int tid  = threadIdx.x;
    const int lane = tid & 31;
    const int wave = tid >> 5;          // 0..7

    const int z  = blockIdx.z;
    const int bb = z / Hz;
    const int hh = z % Hz;
    const unsigned short* Ab = A  + bb * sAb + hh * sAh;
    const unsigned short* Bb = Bt + bb * sBb + hh * sBh;
    const long long coff = bb * sCb + hh * sCh;

    const int row0 = blockIdx.y * BM;
    const int col0 = blockIdx.x * BN;

    // Per-thread async chunk addresses (8 consecutive K = 16B per chunk).
    // A: 256x32 = 1024 chunks -> 4 per thread.
    const unsigned short* gA[4];
    unsigned ldsA[4];
    #pragma unroll
    for (int r = 0; r < 4; ++r) {
        int c = tid + r * 256;
        int i = c >> 2;            // row (4 chunks per 32-K row)
        int k = (c & 3) << 3;      // k, multiple of 8
        gA[r]  = Ab + (long long)(row0 + i) * lda + k;
        ldsA[r] = lds_addr_of(&Asl[i * BK + k]);
    }
    // B: 32x64 = 256 chunks -> 1 per thread, written in fragment order:
    // (k,n) -> [n>>4][(k>>4)*16 + (n&15)][k&15]; 8 consecutive K contiguous.
    const unsigned short* gB;
    unsigned ldsB;
    {
        int n = tid >> 2;
        int k = (tid & 3) << 3;
        gB = Bb + (long long)(col0 + n) * ldb + k;
        int fl = ((k >> 4) << 4) + (n & 15);
        ldsB = lds_addr_of(&Bswz[((n >> 4) * 32 + fl) * 16 + (k & 15)]);
    }

    v8f acc[2][4] = { { {}, {}, {}, {} }, { {}, {}, {}, {} } };

    for (int k0 = 0; k0 < Kd; k0 += BK) {
        // ---- async stage: 5 global_load_async_to_lds_b128 per thread-slot
        #pragma unroll
        for (int r = 0; r < 4; ++r) async_copy_b128(ldsA[r], gA[r] + k0);
        async_copy_b128(ldsB, gB + k0);
        wait_asynccnt0();
        __syncthreads();

        // ---- A fragments (contiguous k = 16*half + s, matches B): 2x ds_load_b128 each.
        v16bf afrag[2];
        #pragma unroll
        for (int m = 0; m < 2; ++m) {
            int rloc = wave * 32 + m * 16 + (lane & 15);
            int hi   = lane >> 4;
            const unsigned short* rp = &Asl[rloc * BK];
            afrag[m] = make_frag(*(const u16x8*)(rp + hi * 16),
                                 *(const u16x8*)(rp + hi * 16 + 8));
        }

        // ---- 4 B fragments x 2 M-subtiles = 8 WMMAs, B reused twice.
        #pragma unroll
        for (int t = 0; t < 4; ++t) {
            const unsigned short* bp = &Bswz[(t * 32 + lane) * 16];
            v16bf bfrag = make_frag(*(const u16x8*)(bp),
                                    *(const u16x8*)(bp + 8));
            acc[0][t] = __builtin_amdgcn_wmma_f32_16x16x32_bf16(
                false, afrag[0], false, bfrag, (short)0, acc[0][t], false, false);
            acc[1][t] = __builtin_amdgcn_wmma_f32_16x16x32_bf16(
                false, afrag[1], false, bfrag, (short)0, acc[1][t], false, false);
        }
        __syncthreads();
    }

    // ---- store: f32 C layout: lane group selects M half, VGPR v -> row g*8+v
    const int cl = lane & 15;
    #pragma unroll
    for (int m = 0; m < 2; ++m) {
        const int rbase = row0 + wave * 32 + m * 16 + (lane >> 4) * 8;
        #pragma unroll
        for (int t = 0; t < 4; ++t) {
            int c = col0 + t * 16 + cl;
            #pragma unroll
            for (int v = 0; v < 8; ++v) {
                long long cidx = coff + (long long)(rbase + v) * ldc + c;
                if constexpr (CF32) ((float*)C)[cidx] = acc[m][t][v];
                else ((unsigned short*)C)[cidx] = f32_to_bf16_bits(acc[m][t][v]);
            }
        }
    }
}

// ---------------------------------------------------------------------------
// One-time conversions.
// ---------------------------------------------------------------------------

// Flat f32 -> bf16, 4 elements/thread (n4 = total/4).
__global__ void conv_bf16(const float* __restrict__ in,
                          unsigned short* __restrict__ out, long long n4)
{
    long long idx = (long long)blockIdx.x * blockDim.x + threadIdx.x;
    if (idx >= n4) return;
    f32x4 v = ((const f32x4*)in)[idx];
    u16x4 h;
    h[0] = f32_to_bf16_bits(v[0]);
    h[1] = f32_to_bf16_bits(v[1]);
    h[2] = f32_to_bf16_bits(v[2]);
    h[3] = f32_to_bf16_bits(v[3]);
    ((u16x4*)out)[idx] = h;
}

// Batched transpose + convert: in [Z][Rr][Cc] f32 -> out [Z][Cc][Rr] bf16.
__global__ void conv_transpose_bf16(const float* __restrict__ in,
                                    unsigned short* __restrict__ out,
                                    int Rr, int Cc, long long total)
{
    long long idx = (long long)blockIdx.x * blockDim.x + threadIdx.x;
    if (idx >= total) return;
    int c = (int)(idx % Cc);
    long long t2 = idx / Cc;
    int r = (int)(t2 % Rr);
    long long zz = t2 / Rr;
    out[(zz * Cc + c) * (long long)Rr + r] = f32_to_bf16_bits(in[idx]);
}

// ---------------------------------------------------------------------------
// RoPE (repeat_interleave(2) tables + rotate_half). Thread owns the pair
// (d, d+half) so in-place is race-free. inv[j'] = 10000^(-2j'/dim), j' = d>>1.
// ---------------------------------------------------------------------------
__device__ __forceinline__ void rope_pair(float x0, float x1, int pos, int j,
                                          int halfD, int D, float* o0, float* o1)
{
    const float LN10K = 9.210340371976184f;  // ln(10000)
    float inv0 = __expf(-((float)((j >> 1) * 2)           / (float)D) * LN10K);
    float inv1 = __expf(-((float)(((j + halfD) >> 1) * 2) / (float)D) * LN10K);
    float s0, c0, s1, c1;
    __sincosf((float)pos * inv0, &s0, &c0);
    __sincosf((float)pos * inv1, &s1, &c1);
    *o0 = x0 * c0 - x1 * s0;   // rot(d<half)  = -x[d+half]
    *o1 = x1 * c1 + x0 * s1;   // rot(d>=half) =  x[d-half]
}

// q_big in-place RoPE on bf16: layout [B,S,H,D], pos = s.
__global__ void rope_q_kernel(unsigned short* __restrict__ q, int S, int H, int D,
                              long long total_pairs)
{
    long long idx = (long long)blockIdx.x * blockDim.x + threadIdx.x;
    if (idx >= total_pairs) return;
    int halfD = D >> 1;
    int j = (int)(idx % halfD);
    long long rest = idx / halfD;          // (b*S + s)*H + h
    long long bs = rest / H;
    int s = (int)(bs % S);
    unsigned short* base = q + rest * (long long)D;
    float o0, o1;
    rope_pair(bf16_bits_to_f32(base[j]), bf16_bits_to_f32(base[j + halfD]),
              s, j, halfD, D, &o0, &o1);
    base[j]         = f32_to_bf16_bits(o0);
    base[j + halfD] = f32_to_bf16_bits(o1);
}

// kv RoPE: f32 [B,L,D] -> bf16 kv_r [B,L,D] AND bf16 kv_rT [B,D,L], pos = l.
__global__ void rope_kv_kernel(const float* __restrict__ kv,
                               unsigned short* __restrict__ out,
                               unsigned short* __restrict__ outT,
                               int L, int D, long long total_pairs)
{
    long long idx = (long long)blockIdx.x * blockDim.x + threadIdx.x;
    if (idx >= total_pairs) return;
    int halfD = D >> 1;
    int j = (int)(idx % halfD);
    long long rest = idx / halfD;          // b*L + l
    long long b = rest / L;
    int l = (int)(rest % L);
    const float* src = kv + rest * (long long)D;
    float o0, o1;
    rope_pair(src[j], src[j + halfD], l, j, halfD, D, &o0, &o1);
    unsigned short h0 = f32_to_bf16_bits(o0);
    unsigned short h1 = f32_to_bf16_bits(o1);
    out[rest * (long long)D + j]         = h0;
    out[rest * (long long)D + j + halfD] = h1;
    outT[(b * D + j)         * (long long)L + l] = h0;
    outT[(b * D + j + halfD) * (long long)L + l] = h1;
}

// ---------------------------------------------------------------------------
// Row softmax on bf16 storage (f32 math). One 256-thread block per row.
// ---------------------------------------------------------------------------
__global__ __launch_bounds__(256)
void softmax_rows(unsigned short* __restrict__ data, int cols)
{
    unsigned short* p = data + (long long)blockIdx.x * cols;
    __shared__ float red[8];
    const int tid = threadIdx.x;

    float m = -INFINITY;
    for (int c = tid; c < cols; c += 256) m = fmaxf(m, bf16_bits_to_f32(p[c]));
    #pragma unroll
    for (int o = 16; o > 0; o >>= 1) m = fmaxf(m, __shfl_xor(m, o, 32));
    if ((tid & 31) == 0) red[tid >> 5] = m;
    __syncthreads();
    if (tid == 0) {
        float v = red[0];
        #pragma unroll
        for (int i = 1; i < 8; ++i) v = fmaxf(v, red[i]);
        red[0] = v;
    }
    __syncthreads();
    m = red[0];
    __syncthreads();

    float s = 0.f;
    for (int c = tid; c < cols; c += 256) {
        float e = __expf(bf16_bits_to_f32(p[c]) - m);
        s += e;
    }
    #pragma unroll
    for (int o = 16; o > 0; o >>= 1) s += __shfl_xor(s, o, 32);
    if ((tid & 31) == 0) red[tid >> 5] = s;
    __syncthreads();
    if (tid == 0) {
        float v = 0.f;
        #pragma unroll
        for (int i = 0; i < 8; ++i) v += red[i];
        red[0] = v;
    }
    __syncthreads();
    float inv = 1.0f / red[0];
    for (int c = tid; c < cols; c += 256)
        p[c] = f32_to_bf16_bits(__expf(bf16_bits_to_f32(p[c]) - m) * inv);
}

// ---------------------------------------------------------------------------
// Host-side orchestration
// ---------------------------------------------------------------------------
static void launch_gemm(hipStream_t st, const unsigned short* A,
                        const unsigned short* Bt, void* C,
                        int M, int N, int Kd, int lda, int ldb, int ldc,
                        long long sAb, long long sAh,
                        long long sBb, long long sBh,
                        long long sCb, long long sCh,
                        int Z, int Hz, int cIsF32)
{
    dim3 g(N / BN, M / BM, Z), b(256);
    if (cIsF32)
        hipLaunchKernelGGL((gemm_bf16_wmma<true>), g, b, 0, st, A, Bt, C, Kd,
                           lda, ldb, ldc, sAb, sAh, sBb, sBh, sCb, sCh, Hz);
    else
        hipLaunchKernelGGL((gemm_bf16_wmma<false>), g, b, 0, st, A, Bt, C, Kd,
                           lda, ldb, ldc, sAb, sAh, sBb, sBh, sCb, sCh, Hz);
}

static void launch_conv(hipStream_t st, const float* in, unsigned short* out,
                        long long n)
{
    long long n4 = n / 4;
    hipLaunchKernelGGL(conv_bf16, dim3((unsigned)((n4 + 255) / 256)), dim3(256),
                       0, st, in, out, n4);
}

static void launch_convT(hipStream_t st, const float* in, unsigned short* out,
                         int Rr, int Cc, long long total)
{
    hipLaunchKernelGGL(conv_transpose_bf16,
                       dim3((unsigned)((total + 255) / 256)), dim3(256),
                       0, st, in, out, Rr, Cc, total);
}

extern "C" void kernel_launch(void* const* d_in, const int* in_sizes, int n_in,
                              void* d_out, int out_size, void* d_ws, size_t ws_size,
                              hipStream_t stream)
{
    (void)in_sizes; (void)n_in; (void)out_size; (void)ws_size;

    const int Bn = 2, S = 1024, L = 1024, DIMQ = 2048, H = 16, Kh = 128;
    const int DC = 512, DCQ = 1536, R = 64;
    const int BS = Bn * S;      // 2048
    const int BL = Bn * L;      // 2048
    const int HK = H * Kh;      // 2048

    const float* hidden  = (const float*)d_in[0];   // [B,S,DIMQ]
    const float* kv_c    = (const float*)d_in[1];   // [B,L,DC]
    const float* Wq      = (const float*)d_in[2];   // [HK,DIMQ]
    const float* w_kc_q  = (const float*)d_in[3];   // [H,Kh,DCQ]
    const float* w_kc_kv = (const float*)d_in[4];   // [H,Kh,DC]
    const float* W_qr    = (const float*)d_in[5];   // [H,DCQ,R]
    const float* W_kr    = (const float*)d_in[6];   // [H,DC,R]
    const float* Wo      = (const float*)d_in[7];   // [DIMQ,HK]
    float* out = (float*)d_out;                     // [B,S,DIMQ]

    // Workspace layout (bf16 = u16 elements). Total ~134M u16 (~268 MB).
    unsigned short* ws = (unsigned short*)d_ws;
    unsigned short* hidden_bf = ws;                                      // BS*DIMQ
    unsigned short* Wq_bf     = hidden_bf + (long long)BS * DIMQ;        // HK*DIMQ
    unsigned short* wkcqT     = Wq_bf     + (long long)HK * DIMQ;        // H*DCQ*Kh
    unsigned short* wqrT      = wkcqT     + (long long)H * DCQ * Kh;     // H*R*DCQ
    unsigned short* wkrT      = wqrT      + (long long)H * R * DCQ;      // H*R*DC
    unsigned short* wkckv_bf  = wkrT      + (long long)H * R * DC;       // H*Kh*DC
    unsigned short* Wo_bf     = wkckv_bf  + (long long)H * Kh * DC;      // DIMQ*HK
    unsigned short* q_hk      = Wo_bf     + (long long)DIMQ * HK;        // BS*HK
    unsigned short* q_big     = q_hk      + (long long)BS * HK;          // BS*H*DCQ
    unsigned short* kv_r      = q_big     + (long long)BS * H * DCQ;     // BL*DC
    unsigned short* kv_rT     = kv_r      + (long long)BL * DC;          // B*DC*L
    unsigned short* q_r       = kv_rT     + (long long)Bn * DC * L;      // BS*H*R
    unsigned short* k_r       = q_r       + (long long)BS * H * R;       // BL*H*R
    unsigned short* scores    = k_r       + (long long)BL * H * R;       // B*H*S*L
    unsigned short* ctx_c     = scores    + (long long)Bn * H * S * L;   // B*H*S*DC
    unsigned short* ctx_lat   = ctx_c     + (long long)Bn * H * S * DC;  // BS*HK

    // ---- one-time conversions / transposes
    launch_conv (stream, hidden,   hidden_bf, (long long)BS * DIMQ);
    launch_conv (stream, Wq,       Wq_bf,     (long long)HK * DIMQ);
    launch_convT(stream, w_kc_q,   wkcqT,  Kh,  DCQ, (long long)H * Kh * DCQ);
    launch_convT(stream, W_qr,     wqrT,   DCQ, R,   (long long)H * DCQ * R);
    launch_convT(stream, W_kr,     wkrT,   DC,  R,   (long long)H * DC * R);
    launch_conv (stream, w_kc_kv,  wkckv_bf, (long long)H * Kh * DC);
    launch_conv (stream, Wo,       Wo_bf,    (long long)DIMQ * HK);

    // ---- kv_r = RoPE(kv_c) (+ transposed copy), pos = l
    {
        long long pairs = (long long)BL * (DC / 2);
        hipLaunchKernelGGL(rope_kv_kernel,
                           dim3((unsigned)((pairs + 255) / 256)), dim3(256),
                           0, stream, kv_c, kv_r, kv_rT, L, DC, pairs);
    }

    // 1) q_hk = hidden @ Wq^T
    launch_gemm(stream, hidden_bf, Wq_bf, q_hk, BS, HK, DIMQ, DIMQ, DIMQ, HK,
                0, 0, 0, 0, 0, 0, 1, 1, 0);

    // 2) q_big[b,s,h,:] = q_hk[b,s,h,:] @ w_kc_q[h]  (Bt = wkcqT [H][DCQ][Kh])
    launch_gemm(stream, q_hk, wkcqT, q_big, BS, DCQ, Kh,
                HK, Kh, H * DCQ,
                0, Kh, 0, (long long)DCQ * Kh, 0, DCQ,
                H, H, 0);

    // 3) RoPE(q_big) in place, pos = s
    {
        long long pairs = (long long)BS * H * (DCQ / 2);
        hipLaunchKernelGGL(rope_q_kernel,
                           dim3((unsigned)((pairs + 255) / 256)), dim3(256),
                           0, stream, q_big, S, H, DCQ, pairs);
    }

    // 5) q_r[b,s,h,:] = q_big[b,s,h,:] @ W_qr[h]  (Bt = wqrT [H][R][DCQ])
    launch_gemm(stream, q_big, wqrT, q_r, BS, R, DCQ,
                H * DCQ, DCQ, H * R,
                0, DCQ, 0, (long long)R * DCQ, 0, R,
                H, H, 0);

    // 6) k_r[b,l,h,:] = kv_r[b,l,:] @ W_kr[h]  (Bt = wkrT [H][R][DC])
    launch_gemm(stream, kv_r, wkrT, k_r, BL, R, DC,
                DC, DC, H * R,
                0, 0, 0, (long long)R * DC, 0, R,
                H, H, 0);

    // 7) scores[b,h] = q_r[b,:,h,:] @ k_r[b,:,h,:]^T  (Bt = k_r, K-contig)
    launch_gemm(stream, q_r, k_r, scores, S, L, R,
                H * R, H * R, L,
                (long long)S * H * R, R,
                (long long)L * H * R, R,
                (long long)H * S * L, (long long)S * L,
                Bn * H, H, 0);

    // 8) softmax over L (rows = B*H*S)
    hipLaunchKernelGGL(softmax_rows, dim3(Bn * H * S), dim3(256), 0, stream,
                       scores, L);

    // 9) ctx_c[b,h] = attn[b,h] @ kv_r[b]  (Bt = kv_rT [B][DC][L])
    launch_gemm(stream, scores, kv_rT, ctx_c, S, DC, L,
                L, L, DC,
                (long long)H * S * L, (long long)S * L,
                (long long)DC * L, 0,
                (long long)H * S * DC, (long long)S * DC,
                Bn * H, H, 0);

    // 10) ctx_lat[b,s,h,:] = ctx_c[b,h,s,:] @ w_kc_kv[h]^T  (Bt = wkckv [H][Kh][DC])
    launch_gemm(stream, ctx_c, wkckv_bf, ctx_lat, S, Kh, DC,
                DC, DC, HK,
                (long long)H * S * DC, (long long)S * DC,
                0, (long long)Kh * DC,
                (long long)S * HK, Kh,
                Bn * H, H, 0);

    // 11) out = ctx_lat @ Wo^T  (f32 store to d_out)
    launch_gemm(stream, ctx_lat, Wo_bf, out, BS, DIMQ, HK, HK, HK, DIMQ,
                0, 0, 0, 0, 0, 0, 1, 1, 1);
}
